// SelfAttention_34368328303353
// MI455X (gfx1250) — matrix operand order repaired and checked by reading, not compiled, not run on hardware
//
#include <hip/hip_runtime.h>

// ---------------------------------------------------------------------------
// Types for CDNA5 WMMA (wave32)
// ---------------------------------------------------------------------------
typedef __attribute__((ext_vector_type(16))) __bf16 v16bf;
typedef __attribute__((ext_vector_type(8)))  float  v8f;
typedef unsigned short u16;

union Frag { uint4 q[2]; v16bf v; };

__device__ __forceinline__ u16 f32_to_bf16(float f) {
    unsigned int u = __builtin_bit_cast(unsigned int, f);
    u += 0x7FFFu + ((u >> 16) & 1u);          // round-to-nearest-even
    return (u16)(u >> 16);
}

// Async global->LDS copy of 16 B per lane (CDNA5, ASYNCcnt-tracked).
__device__ __forceinline__ void async_copy_b128(unsigned lds_byte_addr,
                                                const void* gaddr) {
    asm volatile("global_load_async_to_lds_b128 %0, %1, off"
                 :: "v"(lds_byte_addr), "v"(gaddr) : "memory");
}
__device__ __forceinline__ void wait_asynccnt0() {
    asm volatile("s_wait_asynccnt 0x0" ::: "memory");
}
// Generic (flat) address of LDS object -> 32-bit LDS byte address
// (ISA: LDS-aperture flat addresses carry the LDS offset in addr[31:0]).
__device__ __forceinline__ unsigned lds_addr_of(const void* p) {
    return (unsigned)(unsigned long long)p;
}

// ---------------------------------------------------------------------------
// fp32 -> bf16 transpose:  dst[c*R + r] = bf16(src[r*C + c])
// ---------------------------------------------------------------------------
__global__ __launch_bounds__(256) void xpose_bf16_k(const float* __restrict__ src,
                                                    u16* __restrict__ dst,
                                                    int R, int C) {
    int idx = blockIdx.x * 256 + threadIdx.x;
    if (idx >= R * C) return;
    int r = idx / C, c = idx - r * C;
    dst[(size_t)c * R + r] = f32_to_bf16(src[idx]);
}

// ---------------------------------------------------------------------------
// LayerNorm row kernel: one 256-thread block per row, D == 768 (3 elems/thread)
// ---------------------------------------------------------------------------
__global__ __launch_bounds__(256) void ln_bf16_k(const float* __restrict__ x,
                                                 const float* __restrict__ g,
                                                 const float* __restrict__ beta,
                                                 u16* __restrict__ out, int D) {
    __shared__ float red[256];
    const int row = blockIdx.x, tid = threadIdx.x;
    const float* xr = x + (size_t)row * D;
    float v[3];
    float s = 0.f;
#pragma unroll
    for (int i = 0; i < 3; ++i) { v[i] = xr[tid + i * 256]; s += v[i]; }
    red[tid] = s; __syncthreads();
    for (int st = 128; st > 0; st >>= 1) { if (tid < st) red[tid] += red[tid + st]; __syncthreads(); }
    const float mu = red[0] / (float)D;
    __syncthreads();
    float s2 = 0.f;
#pragma unroll
    for (int i = 0; i < 3; ++i) { float d = v[i] - mu; s2 += d * d; }
    red[tid] = s2; __syncthreads();
    for (int st = 128; st > 0; st >>= 1) { if (tid < st) red[tid] += red[tid + st]; __syncthreads(); }
    const float inv = rsqrtf(red[0] / (float)D + 1e-8f);
#pragma unroll
    for (int i = 0; i < 3; ++i) {
        int c = tid + i * 256;
        out[(size_t)row * D + c] = f32_to_bf16((v[i] - mu) * inv * g[c] + beta[c]);
    }
}

// ---------------------------------------------------------------------------
// Row softmax, in place: reads 4096 fp32, writes 4096 bf16 over the same row.
// ---------------------------------------------------------------------------
__global__ __launch_bounds__(256) void softmax_rows_k(float* __restrict__ sc, int S) {
    __shared__ float red[256];
    const int row = blockIdx.x, tid = threadIdx.x;
    float* rp = sc + (size_t)row * S;
    float v[16];
    float m = -3.4e38f;
#pragma unroll
    for (int i = 0; i < 16; ++i) { v[i] = rp[tid + i * 256]; m = fmaxf(m, v[i]); }
    red[tid] = m; __syncthreads();
    for (int st = 128; st > 0; st >>= 1) { if (tid < st) red[tid] = fmaxf(red[tid], red[tid + st]); __syncthreads(); }
    m = red[0]; __syncthreads();
    float s = 0.f;
#pragma unroll
    for (int i = 0; i < 16; ++i) { v[i] = expf(v[i] - m); s += v[i]; }
    red[tid] = s; __syncthreads();
    for (int st = 128; st > 0; st >>= 1) { if (tid < st) red[tid] += red[tid + st]; __syncthreads(); }
    const float inv = 1.f / red[0];
    __syncthreads();                      // all fp32 reads done before bf16 overwrite
    u16* hp = (u16*)rp;
#pragma unroll
    for (int i = 0; i < 16; ++i) hp[tid + i * 256] = f32_to_bf16(v[i] * inv);
}

// ---------------------------------------------------------------------------
// Async-staged bf16 WMMA GEMM:  C[M,N] = A[M,K] (lda) x Bt[N,K]^T (ldb)
// 256 threads = 8 waves; block tile 128x128; wave tile 64x32 (4x2 wmma frags).
// K-step 32, double-buffered in LDS (2x28 KB), filled by
// GLOBAL_LOAD_ASYNC_TO_LDS_B128 (ASYNCcnt) issued one slab ahead.
// M, N multiples of 128; K multiple of 32.
// OUTM: 0 = fp32 out, 1 = bf16 out, 2 = bf16 transposed-per-batch out.
// ---------------------------------------------------------------------------
template <int OUTM, bool HAS_BIAS, bool DO_GELU, bool HAS_RESID, bool IS_SCORES>
__global__ __launch_bounds__(256) void gemm_bf16_k(
    const u16* __restrict__ A, int lda,
    const u16* __restrict__ Bt, int ldb, int K,
    float* __restrict__ outF, u16* __restrict__ outH, int ldo,
    const float* __restrict__ bias,
    const float* __restrict__ resid, int ldr,
    const int* __restrict__ mask, int mask_ld, float scale,
    int tRows, long long tBatch, int ldoT) {

    // LDS row stride 56 u16 = 112 B: keeps b128 alignment, and 28 words mod 64
    // banks has period 16 -> the 16 rows/cols of a fragment hit distinct banks.
    constexpr int LS = 56;
    __shared__ __align__(16) u16 Ash[2][128 * LS];
    __shared__ __align__(16) u16 Bsh[2][128 * LS];

    const int tid  = threadIdx.x;
    const int lane = tid & 31;
    const int w    = tid >> 5;
    const int wm   = w >> 2, wn = w & 3;
    const int r    = lane & 15, hi = lane >> 4;
    const int tileM = blockIdx.y * 128, tileN = blockIdx.x * 128;

    // Staging map: 512 chunks of 16 B per tile (128 rows x 4 segs), 2 per thread.
    const int c0 = tid * 2, c1 = tid * 2 + 1;
    const int ar0 = c0 >> 2, as0 = (c0 & 3) * 8;
    const int ar1 = c1 >> 2, as1 = (c1 & 3) * 8;
    const u16* gA0 = A  + (size_t)(tileM + ar0) * lda + as0;
    const u16* gA1 = A  + (size_t)(tileM + ar1) * lda + as1;
    const u16* gB0 = Bt + (size_t)(tileN + ar0) * ldb + as0;
    const u16* gB1 = Bt + (size_t)(tileN + ar1) * ldb + as1;

    // LDS byte addresses (per double-buffer half) for the async copies.
    unsigned lA0[2], lA1[2], lB0[2], lB1[2];
#pragma unroll
    for (int p = 0; p < 2; ++p) {
        lA0[p] = lds_addr_of(&Ash[p][ar0 * LS + as0]);
        lA1[p] = lds_addr_of(&Ash[p][ar1 * LS + as1]);
        lB0[p] = lds_addr_of(&Bsh[p][ar0 * LS + as0]);
        lB1[p] = lds_addr_of(&Bsh[p][ar1 * LS + as1]);
    }

    v8f acc[4][2];
#pragma unroll
    for (int i = 0; i < 4; ++i)
#pragma unroll
        for (int j = 0; j < 2; ++j)
#pragma unroll
            for (int e = 0; e < 8; ++e) acc[i][j][e] = 0.0f;

    // Prologue: async-stage K-slab 0 into buffer 0.
    async_copy_b128(lA0[0], gA0);
    async_copy_b128(lA1[0], gA1);
    async_copy_b128(lB0[0], gB0);
    async_copy_b128(lB1[0], gB1);

    int p = 0;
    for (int k0 = 0; k0 < K; k0 += 32, p ^= 1) {
        wait_asynccnt0();       // this wave's copies into buffer p have landed
        __syncthreads();        // ... and every other wave's too; buffer p^1 free

        // Stage the next K-slab into the other buffer while we compute,
        // and gfx1250 global_prefetch the slab after next into L2.
        if (k0 + 32 < K) {
            async_copy_b128(lA0[p ^ 1], gA0 + k0 + 32);
            async_copy_b128(lA1[p ^ 1], gA1 + k0 + 32);
            async_copy_b128(lB0[p ^ 1], gB0 + k0 + 32);
            async_copy_b128(lB1[p ^ 1], gB1 + k0 + 32);
            if (k0 + 64 < K) {
                __builtin_prefetch(gA0 + k0 + 64, 0, 1);
                __builtin_prefetch(gB0 + k0 + 64, 0, 1);
            }
        }

        // B fragments from LDS: lane r = column; K contiguous [hi*16, +16).
        Frag bfr[2];
#pragma unroll
        for (int j = 0; j < 2; ++j) {
            const int off = (wn * 32 + j * 16 + r) * LS + hi * 16;
            bfr[j].q[0] = *(const uint4*)&Bsh[p][off];
            bfr[j].q[1] = *(const uint4*)&Bsh[p][off + 8];
        }
#pragma unroll
        for (int i = 0; i < 4; ++i) {
            // A fragment: lane r = row; K contiguous [hi*16, +16).
            Frag a;
            const int off = (wm * 64 + i * 16 + r) * LS + hi * 16;
            a.q[0] = *(const uint4*)&Ash[p][off];
            a.q[1] = *(const uint4*)&Ash[p][off + 8];
#pragma unroll
            for (int j = 0; j < 2; ++j)
                acc[i][j] = __builtin_amdgcn_wmma_f32_16x16x32_bf16(
                    false, a.v, false, bfr[j].v, (short)0, acc[i][j], false, false);
        }
    }

    // Epilogue.  C layout: lane r = column; VGPR e = row (+8 for upper lanes).
    const int m0 = tileM + wm * 64;
    const int n0 = tileN + wn * 32;
#pragma unroll
    for (int i = 0; i < 4; ++i) {
#pragma unroll
        for (int j = 0; j < 2; ++j) {
            const int col = n0 + j * 16 + r;
            const float bv = HAS_BIAS ? bias[col] : 0.0f;
#pragma unroll
            for (int e = 0; e < 8; ++e) {
                const int row = m0 + i * 16 + hi * 8 + e;
                float val = acc[i][j][e];
                if (IS_SCORES) {
                    const int mk = mask[(size_t)row * mask_ld + col];
                    val = (mk != 0) ? val * scale : -3.0e38f;
                }
                if (HAS_BIAS)  val += bv;
                if (DO_GELU)   val = 0.5f * val * (1.0f + erff(val * 0.70710678118654752f));
                if (HAS_RESID) val += resid[(size_t)row * ldr + col];
                if (OUTM == 0) {
                    outF[(size_t)row * ldo + col] = val;
                } else if (OUTM == 1) {
                    outH[(size_t)row * ldo + col] = f32_to_bf16(val);
                } else {
                    const int bb = row / tRows;
                    const int ss = row - bb * tRows;
                    outH[(size_t)bb * tBatch + (size_t)col * ldoT + ss] = f32_to_bf16(val);
                }
            }
        }
    }
}

// ---------------------------------------------------------------------------
// Host-side orchestration
// ---------------------------------------------------------------------------
extern "C" void kernel_launch(void* const* d_in, const int* in_sizes, int n_in,
                              void* d_out, int out_size, void* d_ws, size_t ws_size,
                              hipStream_t stream) {
    (void)in_sizes; (void)n_in; (void)out_size; (void)ws_size;
    const int B = 2, S = 4096, D = 768, F = 3072;
    const int M = B * S;

    const float* x     = (const float*)d_in[0];
    const int*   mask  = (const int*)  d_in[1];
    const float* ln1_g = (const float*)d_in[2];
    const float* ln1_b = (const float*)d_in[3];
    const float* wq    = (const float*)d_in[4];
    const float* wk    = (const float*)d_in[5];
    const float* wv    = (const float*)d_in[6];
    const float* wo    = (const float*)d_in[7];
    const float* bo    = (const float*)d_in[8];
    const float* ln2_g = (const float*)d_in[9];
    const float* ln2_b = (const float*)d_in[10];
    const float* w1    = (const float*)d_in[11];
    const float* b1    = (const float*)d_in[12];
    const float* w2    = (const float*)d_in[13];
    const float* b2    = (const float*)d_in[14];
    float* out = (float*)d_out;

    // Workspace carve-up (256 B aligned).
    char*  ws  = (char*)d_ws;
    size_t off = 0;
    auto carve = [&](size_t bytes) -> void* {
        void* p = ws + off;
        off = (off + bytes + 255) & ~(size_t)255;
        return p;
    };
    u16* wqT   = (u16*)carve((size_t)D * D * 2);
    u16* wkT   = (u16*)carve((size_t)D * D * 2);
    u16* wvT   = (u16*)carve((size_t)D * D * 2);
    u16* woT   = (u16*)carve((size_t)D * D * 2);
    u16* w1T   = (u16*)carve((size_t)F * D * 2);     // [F][D]
    u16* w2T   = (u16*)carve((size_t)D * F * 2);     // [D][F]
    u16* xn    = (u16*)carve((size_t)M * D * 2);     // LN output (reused for LN2)
    u16* Qb    = (u16*)carve((size_t)M * D * 2);     // Q, later reused as attn_out
    u16* Kb    = (u16*)carve((size_t)M * D * 2);
    u16* Vt    = (u16*)carve((size_t)B * D * S * 2); // [B][D][S]
    u16* h     = (u16*)carve((size_t)M * F * 2);
    float* scr = (float*)carve((size_t)S * S * 4);   // per-batch scores (L2-resident)

    // 1. Weight conversion / transpose to bf16 [N][K].
    {
        int n = D * D;
        xpose_bf16_k<<<(n + 255) / 256, 256, 0, stream>>>(wq, wqT, D, D);
        xpose_bf16_k<<<(n + 255) / 256, 256, 0, stream>>>(wk, wkT, D, D);
        xpose_bf16_k<<<(n + 255) / 256, 256, 0, stream>>>(wv, wvT, D, D);
        xpose_bf16_k<<<(n + 255) / 256, 256, 0, stream>>>(wo, woT, D, D);
        int n1 = D * F;
        xpose_bf16_k<<<(n1 + 255) / 256, 256, 0, stream>>>(w1, w1T, D, F); // -> [F][D]
        xpose_bf16_k<<<(n1 + 255) / 256, 256, 0, stream>>>(w2, w2T, F, D); // -> [D][F]
    }

    // 2. LN1 -> xn (bf16).
    ln_bf16_k<<<M, 256, 0, stream>>>(x, ln1_g, ln1_b, xn, D);

    // 3. QKV projections (V stored transposed per batch: [B][D][S]).
    dim3 gDM(D / 128, M / 128);
    gemm_bf16_k<1, false, false, false, false><<<gDM, 256, 0, stream>>>(
        xn, D, wqT, D, D, nullptr, Qb, D, nullptr, nullptr, 0, nullptr, 0, 0.f, 0, 0, 0);
    gemm_bf16_k<1, false, false, false, false><<<gDM, 256, 0, stream>>>(
        xn, D, wkT, D, D, nullptr, Kb, D, nullptr, nullptr, 0, nullptr, 0, 0.f, 0, 0, 0);
    gemm_bf16_k<2, false, false, false, false><<<gDM, 256, 0, stream>>>(
        xn, D, wvT, D, D, nullptr, Vt, 0, nullptr, nullptr, 0, nullptr, 0, 0.f,
        S, (long long)D * S, S);

    // 4. Attention, one batch at a time (score scratch reused; lives in L2).
    const float scale = 1.0f / sqrtf((float)D);
    for (int b = 0; b < B; ++b) {
        const size_t sd = (size_t)b * S * D;
        // scores = scale * Q Kt, masked  (Bt = K rows: Bt[key][d])
        dim3 gSS(S / 128, S / 128);
        gemm_bf16_k<0, false, false, false, true><<<gSS, 256, 0, stream>>>(
            Qb + sd, D, Kb + sd, D, D, scr, nullptr, S,
            nullptr, nullptr, 0, mask, S, scale, 0, 0, 0);
        // softmax rows, rewrite in place as bf16 P (row stride S fp32 = 2S bf16)
        softmax_rows_k<<<S, 256, 0, stream>>>(scr, S);
        // attn_out = P @ V   (A = P bf16 lda=2S; Bt = Vt[d][key] ldb=S) -> reuse Qb
        dim3 gPV(D / 128, S / 128);
        gemm_bf16_k<1, false, false, false, false><<<gPV, 256, 0, stream>>>(
            (const u16*)scr, 2 * S, Vt + (size_t)b * D * S, S, S,
            nullptr, Qb + sd, D, nullptr, nullptr, 0, nullptr, 0, 0.f, 0, 0, 0);
    }

    // 5. x1 = x + attn_out @ wo + bo  -> d_out (fp32)
    gemm_bf16_k<0, true, false, true, false><<<gDM, 256, 0, stream>>>(
        Qb, D, woT, D, D, out, nullptr, D, bo, x, D, nullptr, 0, 0.f, 0, 0, 0);

    // 6. LN2(d_out) -> xn (bf16, reuse)
    ln_bf16_k<<<M, 256, 0, stream>>>(out, ln2_g, ln2_b, xn, D);

    // 7. h = gelu(xn @ w1 + b1)  (bf16)
    dim3 gFM(F / 128, M / 128);
    gemm_bf16_k<1, true, true, false, false><<<gFM, 256, 0, stream>>>(
        xn, D, w1T, D, D, nullptr, h, F, b1, nullptr, 0, nullptr, 0, 0.f, 0, 0, 0);

    // 8. out = x1 + h @ w2 + b2   (residual read of d_out, write d_out in place)
    gemm_bf16_k<0, true, false, true, false><<<gDM, 256, 0, stream>>>(
        h, F, w2T, F, F, out, nullptr, D, b2, out, D, nullptr, 0, 0.f, 0, 0, 0);
}
